// EdgeGrasp_33775622815779
// MI455X (gfx1250) — matrix operand order repaired and checked by reading, not compiled, not run on hardware
//
#include <hip/hip_runtime.h>
#include <hip/hip_bf16.h>
#include <math.h>

// ---------------------------------------------------------------------------
// EdgeGrasp forward on MI455X (gfx1250): bf16 WMMA (fp32 accum) everywhere.
// ---------------------------------------------------------------------------

typedef __bf16 bf16;
typedef __attribute__((ext_vector_type(16))) __bf16 v16bf;
typedef __attribute__((ext_vector_type(8)))  float  v8f;

#define BB 2
#define NNPTS 4096
#define AA 16
#define MMPTS 512
#define KNN 16
#define CCON 32
#define PP (BB*AA*MMPTS)   /* 16384 per-point rows  */
#define BAD (BB*AA)        /* 32 (b,a) segments     */

// Output layout (fp32), matching tuple (loss, trans, pred, gt, appr_pts):
#define OUT_LOSS 0
#define OUT_TRANS 1                      /* 2*16*32*16 = 16384 */
#define OUT_PRED (OUT_TRANS + 16384)     /* 16385 */
#define OUT_GT   (OUT_PRED + 1024)       /* 17409 */
#define OUT_APPR (OUT_GT + 1024)         /* 18433 (+96 = 18529) */

__device__ __forceinline__ v8f v8f_zero() {
  v8f z = {0.f,0.f,0.f,0.f,0.f,0.f,0.f,0.f};
  return z;
}

__device__ __forceinline__ v8f wmma_bf16(v16bf a, v16bf b, v8f c) {
  // D(16x16 f32) = A(16x32 bf16) * B(32x16 bf16) + C
  return __builtin_amdgcn_wmma_f32_16x16x32_bf16(false, a, false, b,
                                                 (short)0, c, false, false);
}

// A fragment (16x32 bf16) per ISA layout:
//  lanes 0-15 (M=lane):    e0..7 -> K kb+0..7,  e8..15 -> K kb+16..23
//  lanes 16-31 (M=lane-16): e0..7 -> K kb+8..15, e8..15 -> K kb+24..31
// Row-major source (stride = pitch elems, mult of 8) -> two 16B loads.
__device__ __forceinline__ v16bf load_a_frag(const bf16* base, long pitch,
                                             int row0, int lane, int kb) {
  int row = row0 + (lane & 15);
  int hi8 = (lane >> 4) * 8;
  union { v16bf v; uint4 q[2]; } u;
  const bf16* p = base + (long)row * pitch + kb + hi8;
  u.q[0] = *(const uint4*)(p);
  u.q[1] = *(const uint4*)(p + 16);
  return u.v;
}

// B fragment (32x16 bf16): lane holds column n, same K order as A
// (lanes 0-15: K kb+0..7, kb+16..23; lanes 16-31: K kb+8..15, kb+24..31). Source stored
// pre-transposed as WT[n][pitch] so this is two 16B loads.
__device__ __forceinline__ v16bf load_b_frag(const bf16* WT, long pitch,
                                             int n0, int lane, int kb) {
  int n = n0 + (lane & 15);
  int hi16 = (lane >> 4) * 8;
  union { v16bf v; uint4 q[2]; } u;
  const bf16* p = WT + (long)n * pitch + kb + hi16;
  u.q[0] = *(const uint4*)(p);
  u.q[1] = *(const uint4*)(p + 16);
  return u.v;
}

// ---------------------------------------------------------------------------
// Weight prep: fp32 [Kin][N] -> bf16 transposed [N][Kp], zero-padded K.
// ---------------------------------------------------------------------------
__global__ void wconvert_kernel(const float* __restrict__ src, bf16* __restrict__ dst,
                                int Kin, int Kp, int N) {
  long i = (long)blockIdx.x * 256 + threadIdx.x;
  if (i >= (long)Kp * N) return;
  int n = (int)(i / Kp);
  int k = (int)(i - (long)n * Kp);
  dst[i] = (k < Kin) ? (bf16)src[(long)k * N + n] : (bf16)0.f;
}

__global__ void zero_loss_kernel(float* out) {
  if (threadIdx.x == 0 && blockIdx.x == 0) out[OUT_LOSS] = 0.f;
}

// ---------------------------------------------------------------------------
// Gathers: ball_pos/ball_nrm (fp32 + bf16 copy of pos), appr_pts (also to out)
// ---------------------------------------------------------------------------
__global__ void gather_kernel(const float* __restrict__ pos, const float* __restrict__ nrm,
                              const int* __restrict__ ball_idx, const int* __restrict__ appr_idx,
                              float* __restrict__ ballpos, float* __restrict__ ballnrm,
                              bf16* __restrict__ ballpos_h,
                              float* __restrict__ appr, float* __restrict__ out_appr) {
  int p = blockIdx.x * 256 + threadIdx.x;
  if (p < PP) {
    int b = p >> 13;                 // / (A*M) = 8192
    int idx = ball_idx[p];
    long s = ((long)b * NNPTS + idx) * 3;
    #pragma unroll
    for (int d = 0; d < 3; ++d) {
      float pv = pos[s + d];
      ballpos[p * 3 + d] = pv;
      ballpos_h[p * 3 + d] = (bf16)pv;
      ballnrm[p * 3 + d] = nrm[s + d];
    }
  }
  if (p < BAD) {
    int b = p >> 4;                  // / A
    int idx = appr_idx[p];
    #pragma unroll
    for (int d = 0; d < 3; ++d) {
      float v = pos[((long)b * NNPTS + idx) * 3 + d];
      appr[p * 3 + d] = v;
      out_appr[p * 3 + d] = v;
    }
  }
}

// ---------------------------------------------------------------------------
// Fused PointNetConv: per point (one wave each):
//   E[16][Kp1] = [h_j | rel_pos | 0-pad]   (K=16 neighbors == WMMA M dim)
//   H = relu(E @ W1 + b1)        (16 x C1)
//   D = H @ W2 + b2              (16 x C2)
//   out = relu(max over the 16 neighbor rows)  -> one C2-row per point
// Block: 256 thr = 8 waves; weights staged ONCE per block in (dynamic) LDS,
// each wave processes 8 points to amortize staging. CDNA5 allows up to
// 320KB LDS per workgroup (c3 needs ~112KB).
// ---------------------------------------------------------------------------
#define PCONV_WAVES 8
#define PCONV_PTS 8

__global__ __launch_bounds__(256)
void pconv_kernel(const bf16* __restrict__ hprev, int sprev, int Cin,
                  int Kp1, int C1, int C2,
                  const bf16* __restrict__ W1T, const bf16* __restrict__ W2T,
                  const float* __restrict__ b1, const float* __restrict__ b2,
                  const float* __restrict__ ballpos, const int* __restrict__ knn,
                  bf16* __restrict__ outCE, int sout, int colOff) {
  extern __shared__ char smem_raw[];
  bf16* sW1 = (bf16*)smem_raw;                         // [C1][Kp1] transposed
  bf16* sW2 = sW1 + C1 * Kp1;                          // [C2][C1]  transposed
  bf16* sEb = sW2 + C2 * C1;                           // 8 waves * 16 * Kp1
  bf16* sHb = sEb + PCONV_WAVES * 16 * Kp1;            // 8 waves * 16 * C1

  int tid = threadIdx.x;
  int w = tid >> 5, lane = tid & 31;

  // ---- stage transposed weights into LDS (vectorized, once per block) ----
  for (int i = tid; i < (C1 * Kp1) >> 3; i += 256)
    ((uint4*)sW1)[i] = ((const uint4*)W1T)[i];
  for (int i = tid; i < (C2 * C1) >> 3; i += 256)
    ((uint4*)sW2)[i] = ((const uint4*)W2T)[i];
  __syncthreads();

  bf16* E = sEb + w * 16 * Kp1;
  bf16* H = sHb + w * 16 * C1;

  for (int it = 0; it < PCONV_PTS; ++it) {
    int p = (blockIdx.x * PCONV_WAVES + w) * PCONV_PTS + it;  // point id < 16384
    int baBase = (p >> 9) << 9;                               // ball base (M=512)

    // ---- build edge features: lanes split columns between halves ----
    {
      int nb = lane & 15;
      int half = lane >> 4;
      int j = knn[(long)p * KNN + nb];
      int gj = baBase + j;
      float rx = ballpos[gj * 3 + 0] - ballpos[p * 3 + 0];
      float ry = ballpos[gj * 3 + 1] - ballpos[p * 3 + 1];
      float rz = ballpos[gj * 3 + 2] - ballpos[p * 3 + 2];
      for (int c = half; c < Kp1; c += 2) {
        float v;
        if (c < Cin)           v = (float)hprev[(long)gj * sprev + c];
        else if (c == Cin)     v = rx;
        else if (c == Cin + 1) v = ry;
        else if (c == Cin + 2) v = rz;
        else                   v = 0.f;
        E[nb * Kp1 + c] = (bf16)v;
      }
    }
    __syncthreads();

    // ---- GEMM1: H = relu(E @ W1 + b1) ----
    int nt1 = C1 >> 4, kt1 = Kp1 >> 5;
    for (int nt = 0; nt < nt1; ++nt) {
      v8f acc = v8f_zero();
      for (int kt = 0; kt < kt1; ++kt) {
        v16bf a = load_a_frag(E, Kp1, 0, lane, kt * 32);
        v16bf b = load_b_frag(sW1, Kp1, nt * 16, lane, kt * 32);
        acc = wmma_bf16(a, b, acc);
      }
      float bias = b1[nt * 16 + (lane & 15)];
      union { v8f v; float f[8]; } u; u.v = acc;
      int rbase = (lane >> 4) * 8;
      int col = nt * 16 + (lane & 15);
      #pragma unroll
      for (int v = 0; v < 8; ++v) {
        float x = u.f[v] + bias;
        H[(rbase + v) * C1 + col] = (bf16)fmaxf(x, 0.f);
      }
    }
    __syncthreads();

    // ---- GEMM2 + bias + neighbor-max + relu ----
    int nt2 = C2 >> 4, kt2 = C1 >> 5;
    for (int nt = 0; nt < nt2; ++nt) {
      v8f acc = v8f_zero();
      for (int kt = 0; kt < kt2; ++kt) {
        v16bf a = load_a_frag(H, C1, 0, lane, kt * 32);
        v16bf b = load_b_frag(sW2, C1, nt * 16, lane, kt * 32);
        acc = wmma_bf16(a, b, acc);
      }
      float bias = b2[nt * 16 + (lane & 15)];
      union { v8f v; float f[8]; } u; u.v = acc;
      float mx = -3.4e38f;
      #pragma unroll
      for (int v = 0; v < 8; ++v) mx = fmaxf(mx, u.f[v] + bias);  // max over half rows
      mx = fmaxf(mx, __shfl_xor(mx, 16, 32));                     // combine halves
      mx = fmaxf(mx, 0.f);                                        // relu(max)
      if (lane < 16)
        outCE[(long)p * sout + colOff + nt * 16 + lane] = (bf16)mx;
    }
    __syncthreads();   // protect E/H reuse next iteration
  }
}

// ---------------------------------------------------------------------------
// Generic row-GEMM: out[16384, N] = act(A[16384, K] @ W[K,N] + b)
// Block 256 = 8 waves; block tile 256 rows x 64 cols. Each wave: 32 rows
// (2 row-tiles) x 64 cols (4 col-tiles) = 8 accumulators. Weight chunk
// (64 x 128, pitch 136 -> 4-bank stride, 16B aligned) staged in LDS and
// shared by all 8 waves. A frags streamed from global; next K-chunk
// prefetched via global_prefetch.
// ---------------------------------------------------------------------------
__global__ __launch_bounds__(256)
void gemm_kernel(const bf16* __restrict__ A, int strideA, int Kp,
                 const bf16* __restrict__ WT, const float* __restrict__ bias,
                 int Nout, bf16* __restrict__ out, int strideOut, int relu) {
  __shared__ __align__(16) bf16 sB[64 * 136];
  const int PITCH = 136;

  int tid = threadIdx.x, w = tid >> 5, lane = tid & 31;
  int rowBase = blockIdx.x * 256 + w * 32;
  int nb = blockIdx.y * 64;

  v8f acc[2][4];
  #pragma unroll
  for (int r = 0; r < 2; ++r)
    #pragma unroll
    for (int nt = 0; nt < 4; ++nt) acc[r][nt] = v8f_zero();

  for (int kc = 0; kc < Kp; kc += 128) {
    int kchunk = Kp - kc; if (kchunk > 128) kchunk = 128;   // Kp mult of 32
    __syncthreads();
    // stage WT[nb..nb+63][kc..kc+kchunk) -> sB (8 bf16 per copy)
    for (int i = tid; i < 64 * 16; i += 256) {
      int n = i >> 4, k = (i & 15) * 8;
      uint4 v;
      if (k < kchunk) v = *(const uint4*)(WT + (long)(nb + n) * Kp + kc + k);
      else            v = make_uint4(0u, 0u, 0u, 0u);
      *(uint4*)(&sB[n * PITCH + k]) = v;
    }
    __syncthreads();

    // prefetch next A K-chunk (speculative; lowers to global_prefetch_b8)
    if (kc + 128 < Kp) {
      __builtin_prefetch(A + (long)(rowBase + (lane & 15)) * strideA + kc + 128, 0, 3);
      __builtin_prefetch(A + (long)(rowBase + 16 + (lane & 15)) * strideA + kc + 128, 0, 3);
    }

    int kts = kchunk >> 5;
    for (int kt = 0; kt < kts; ++kt) {
      v16bf a0 = load_a_frag(A, strideA, rowBase,      lane, kc + kt * 32);
      v16bf a1 = load_a_frag(A, strideA, rowBase + 16, lane, kc + kt * 32);
      #pragma unroll
      for (int nt = 0; nt < 4; ++nt) {
        v16bf b = load_b_frag(sB, PITCH, nt * 16, lane, kt * 32);
        acc[0][nt] = wmma_bf16(a0, b, acc[0][nt]);
        acc[1][nt] = wmma_bf16(a1, b, acc[1][nt]);
      }
    }
  }

  int rb = (lane >> 4) * 8;
  #pragma unroll
  for (int r = 0; r < 2; ++r) {
    #pragma unroll
    for (int nt = 0; nt < 4; ++nt) {
      int col = nb + nt * 16 + (lane & 15);
      float bv = bias[col];
      union { v8f v; float f[8]; } u; u.v = acc[r][nt];
      #pragma unroll
      for (int v = 0; v < 8; ++v) {
        float x = u.f[v] + bv;
        if (relu) x = fmaxf(x, 0.f);
        out[(long)(rowBase + r * 16 + rb + v) * strideOut + col] = (bf16)x;
      }
    }
  }
}

// max over the 512 rows of each (b,a) segment
__global__ void segmax_kernel(const bf16* __restrict__ in, int F, bf16* __restrict__ out) {
  int col = blockIdx.y * 256 + threadIdx.x;
  int seg = blockIdx.x;
  float mx = -3.4e38f;
  const bf16* p = in + (long)seg * 512 * F + col;
  for (int r = 0; r < 512; ++r) mx = fmaxf(mx, (float)p[(long)r * F]);
  out[(long)seg * F + col] = (bf16)mx;
}

// out[row, 0:224] = CE[row], out[row, 224:] = seg[row/512] (broadcast concat)
__global__ void concat_kernel(const bf16* __restrict__ CE, const bf16* __restrict__ seg,
                              int segF, bf16* __restrict__ out, int strideOut, long total) {
  long i = (long)blockIdx.x * 256 + threadIdx.x;
  if (i >= total) return;
  int row = (int)(i / strideOut);
  int c = (int)(i - (long)row * strideOut);
  out[i] = (c < 224) ? CE[(long)row * 224 + c]
                     : seg[(long)(row >> 9) * segF + (c - 224)];
}

// cls4: 128 -> 1 dot product per point
__global__ void cls4_kernel(const bf16* __restrict__ in, const bf16* __restrict__ w,
                            const float* __restrict__ b, float* __restrict__ logits) {
  int p = blockIdx.x * 256 + threadIdx.x;
  if (p >= PP) return;
  const bf16* r = in + (long)p * 128;
  float s = 0.f;
  #pragma unroll 8
  for (int c = 0; c < 128; ++c) s += (float)r[c] * (float)w[c];
  logits[p] = s + b[0];
}

__device__ __forceinline__ void norm3(float& x, float& y, float& z) {
  float n = sqrtf(x * x + y * y + z * z);
  float r = 1.f / fmaxf(n, 1e-12f);
  x *= r; y *= r; z *= r;
}

// ---------------------------------------------------------------------------
// Tail: contact gather, BCE loss (sum of per-(b,a) mean over contacts),
// grasp 4x4 frames, output packing.
// ---------------------------------------------------------------------------
__global__ void tail_kernel(const float* __restrict__ logits, const int* __restrict__ contact_idx,
                            const float* __restrict__ gt,
                            const float* __restrict__ ballpos, const float* __restrict__ ballnrm,
                            const float* __restrict__ appr, float* __restrict__ out) {
  int t = blockIdx.x * 256 + threadIdx.x;
  if (t >= BAD * CCON) return;
  int ba = t >> 5;
  int cidx = contact_idx[t];
  int gp = ba * 512 + cidx;

  // loss + predictions
  float z = logits[gp];
  float y = gt[t];
  float bce = fmaxf(z, 0.f) - z * y + log1pf(expf(-fabsf(z)));
  atomicAdd(&out[OUT_LOSS], bce * (1.0f / CCON));
  out[OUT_PRED + t] = z;
  out[OUT_GT + t] = y;

  // grasp transform
  float cpx = ballpos[gp * 3 + 0], cpy = ballpos[gp * 3 + 1], cpz = ballpos[gp * 3 + 2];
  float cnx = ballnrm[gp * 3 + 0], cny = ballnrm[gp * 3 + 1], cnz = ballnrm[gp * 3 + 2];
  norm3(cnx, cny, cnz);
  float apx = appr[ba * 3 + 0], apy = appr[ba * 3 + 1], apz = appr[ba * 3 + 2];
  float rlx = apx - cpx, rly = apy - cpy, rlz = apz - cpz;
  float rnx = rlx, rny = rly, rnz = rlz;
  norm3(rnx, rny, rnz);
  // x_axis = normalize(cross(cnrm, rel_n))
  float xx = cny * rnz - cnz * rny;
  float xy = cnz * rnx - cnx * rnz;
  float xz = cnx * rny - cny * rnx;
  norm3(xx, xy, xz);
  // appr_axis = -normalize(cross(x_axis, cnrm))
  float ax = xy * cnz - xz * cny;
  float ay = xz * cnx - xx * cnz;
  float az = xx * cny - xy * cnx;
  norm3(ax, ay, az);
  ax = -ax; ay = -ay; az = -az;
  float dot = -(rlx * ax + rly * ay + rlz * az);
  float shift = 0.065f + dot;                       // GRIPPER_DEPTH = 0.072-0.007
  float gx = apx - shift * ax, gy = apy - shift * ay, gz = apz - shift * az;

  float* T = out + OUT_TRANS + (long)t * 16;
  T[0] = cnx; T[1] = xx; T[2] = ax; T[3] = gx;
  T[4] = cny; T[5] = xy; T[6] = ay; T[7] = gy;
  T[8] = cnz; T[9] = xz; T[10] = az; T[11] = gz;
  T[12] = 0.f; T[13] = 0.f; T[14] = 0.f; T[15] = 1.f;
}

// ---------------------------------------------------------------------------
// Host orchestration
// ---------------------------------------------------------------------------
extern "C" void kernel_launch(void* const* d_in, const int* in_sizes, int n_in,
                              void* d_out, int out_size, void* d_ws, size_t ws_size,
                              hipStream_t stream) {
  const float* pos = (const float*)d_in[0];
  const float* nrm = (const float*)d_in[1];
  const int* appr_idx = (const int*)d_in[2];
  const int* ball_idx = (const int*)d_in[3];
  const int* knn_idx = (const int*)d_in[4];
  const int* contact_idx = (const int*)d_in[5];
  const float* success_gt = (const float*)d_in[6];
  float* out = (float*)d_out;

  // layer table: {Kin, Kp (pad32), N}; params at d_in[7+2l] (W), d_in[8+2l] (b)
  struct L { int Kin, Kp, N; };
  const L ls[15] = {
    {6,32,32},{32,32,32},{35,64,64},{64,64,64},{67,96,128},{128,128,128},
    {224,224,256},{256,256,512},{512,512,512},{736,736,1024},{1024,1024,1024},
    {1248,1248,512},{512,512,256},{256,256,128},{128,128,1}
  };

  char* ws = (char*)d_ws;
  size_t off = 0;
  auto alloc = [&](size_t bytes) -> char* {
    char* r = ws + off;
    off = (off + bytes + 255) & ~(size_t)255;
    return r;
  };

  bf16* wt[15];
  for (int l = 0; l < 15; ++l) wt[l] = (bf16*)alloc((size_t)ls[l].Kp * ls[l].N * 2);
  float* ballpos   = (float*)alloc((size_t)PP * 3 * 4);
  float* ballnrm   = (float*)alloc((size_t)PP * 3 * 4);
  bf16*  ballpos_h = (bf16*) alloc((size_t)PP * 3 * 2);
  float* appr      = (float*)alloc((size_t)BAD * 3 * 4);
  bf16*  CE        = (bf16*) alloc((size_t)PP * 224 * 2);
  bf16*  buf1      = (bf16*) alloc((size_t)PP * 1024 * 2);
  bf16*  buf2      = (bf16*) alloc((size_t)PP * 1024 * 2);
  bf16*  bufBig    = (bf16*) alloc((size_t)PP * 1248 * 2);
  bf16*  gmax      = (bf16*) alloc((size_t)BAD * 512 * 2);
  bf16*  glob      = (bf16*) alloc((size_t)BAD * 1024 * 2);
  float* logits    = (float*)alloc((size_t)PP * 4);

  zero_loss_kernel<<<1, 64, 0, stream>>>(out);

  for (int l = 0; l < 15; ++l) {
    long tot = (long)ls[l].Kp * ls[l].N;
    wconvert_kernel<<<(int)((tot + 255) / 256), 256, 0, stream>>>(
        (const float*)d_in[7 + 2 * l], wt[l], ls[l].Kin, ls[l].Kp, ls[l].N);
  }

  gather_kernel<<<PP / 256, 256, 0, stream>>>(pos, nrm, ball_idx, appr_idx,
      ballpos, ballnrm, ballpos_h, appr, out + OUT_APPR);

  // --- PointNetSimple: 3 fused edge-conv layers writing into CE[16384,224] ---
  const int pconv_grid = PP / (PCONV_WAVES * PCONV_PTS);   // 256 blocks
  auto pconv_smem = [&](int Kp1, int C1, int C2) -> size_t {
    return (size_t)(C1 * Kp1 + C2 * C1 +
                    PCONV_WAVES * 16 * Kp1 + PCONV_WAVES * 16 * C1) * 2;
  };
  pconv_kernel<<<pconv_grid, 256, pconv_smem(32, 32, 32), stream>>>(
      ballpos_h, 3, 3, 32, 32, 32, wt[0], wt[1],
      (const float*)d_in[8], (const float*)d_in[10],
      ballpos, knn_idx, CE, 224, 0);
  pconv_kernel<<<pconv_grid, 256, pconv_smem(64, 64, 64), stream>>>(
      CE, 224, 32, 64, 64, 64, wt[2], wt[3],
      (const float*)d_in[12], (const float*)d_in[14],
      ballpos, knn_idx, CE, 224, 32);
  pconv_kernel<<<pconv_grid, 256, pconv_smem(96, 128, 128), stream>>>(
      CE + 32, 224, 64, 96, 128, 128, wt[4], wt[5],
      (const float*)d_in[16], (const float*)d_in[18],
      ballpos, knn_idx, CE, 224, 96);

  auto gemm = [&](const bf16* Ap, int sA, int Kp, int li, bf16* Op, int sO, int relu) {
    dim3 grid(PP / 256, ls[li].N / 64);
    gemm_kernel<<<grid, 256, 0, stream>>>(Ap, sA, Kp, wt[li],
        (const float*)d_in[8 + 2 * li], ls[li].N, Op, sO, relu);
  };

  // --- GlobalEmdModel ---
  gemm(CE, 224, 224, 6, buf1, 256, 1);        // g1_1 relu
  gemm(buf1, 256, 256, 7, buf2, 512, 1);      // g1_2 relu
  gemm(buf2, 512, 512, 8, buf1, 512, 0);      // g1_3
  segmax_kernel<<<dim3(BAD, 2), 256, 0, stream>>>(buf1, 512, gmax);
  { long tot = (long)PP * 736;
    concat_kernel<<<(int)((tot + 255) / 256), 256, 0, stream>>>(CE, gmax, 512, bufBig, 736, tot); }
  gemm(bufBig, 736, 736, 9, buf1, 1024, 1);   // g2_1 relu
  gemm(buf1, 1024, 1024, 10, buf2, 1024, 0);  // g2_2
  segmax_kernel<<<dim3(BAD, 4), 256, 0, stream>>>(buf2, 1024, glob);

  // --- Classifier ---
  { long tot = (long)PP * 1248;
    concat_kernel<<<(int)((tot + 255) / 256), 256, 0, stream>>>(CE, glob, 1024, bufBig, 1248, tot); }
  gemm(bufBig, 1248, 1248, 11, buf1, 512, 1); // cls1 relu
  gemm(buf1, 512, 512, 12, buf2, 256, 1);     // cls2 relu
  gemm(buf2, 256, 256, 13, buf1, 128, 1);     // cls3 relu
  cls4_kernel<<<PP / 256, 256, 0, stream>>>(buf1, wt[14], (const float*)d_in[36], logits);

  // --- contacts: loss + grasp transforms + outputs ---
  tail_kernel<<<4, 256, 0, stream>>>(logits, contact_idx, success_gt,
      ballpos, ballnrm, appr, out);
}
